// SelfAttention_66056597012775
// MI455X (gfx1250) — matrix operand order repaired and checked by reading, not compiled, not run on hardware
//
#include <hip/hip_runtime.h>
#include <hip/hip_bf16.h>

// ---------------------------------------------------------------------------
// CDNA5 (gfx1250) self-attention block.
// bf16 WMMA (v_wmma_f32_16x16x32_bf16) for all three matmul stages.
// DPP16 butterfly reductions; async global->LDS copies; TDM tensor DMA.
// ---------------------------------------------------------------------------

typedef __attribute__((ext_vector_type(16))) __bf16 bf16x16;
typedef __attribute__((ext_vector_type(8)))  float  f32x8;

union FragAB {
    bf16x16 v;
    uint4   q[2];
};

__device__ __forceinline__ f32x8 wmma_bf16(bf16x16 a, bf16x16 b, f32x8 c) {
    // (neg_a, A, neg_b, B, c_mod, C, reuse_a, reuse_b)
    return __builtin_amdgcn_wmma_f32_16x16x32_bf16(false, a, false, b,
                                                   (short)0, c, false, false);
}

__device__ __forceinline__ unsigned short f2bf(float f) {
    unsigned u = __builtin_bit_cast(unsigned, f);
    return (unsigned short)((u + 0x7FFFu + ((u >> 16) & 1u)) >> 16);
}
__device__ __forceinline__ unsigned pack2(float a, float b) {
    return (unsigned)f2bf(a) | ((unsigned)f2bf(b) << 16);
}

// ---- DPP16 butterfly reductions (within each 16-lane half-row) ------------
#if __has_builtin(__builtin_amdgcn_mov_dpp)
template <int CTRL>
__device__ __forceinline__ float xdpp(float v) {
    return __builtin_bit_cast(float,
        __builtin_amdgcn_mov_dpp(__builtin_bit_cast(int, v), CTRL, 0xF, 0xF, false));
}
__device__ __forceinline__ float rowmax16(float v) {
    v = fmaxf(v, xdpp<0xB1>(v));    // quad_perm(1,0,3,2)  : xor 1
    v = fmaxf(v, xdpp<0x4E>(v));    // quad_perm(2,3,0,1)  : xor 2
    v = fmaxf(v, xdpp<0x141>(v));   // row_half_mirror     : xor 4 (quads uniform)
    v = fmaxf(v, xdpp<0x140>(v));   // row_mirror          : xor 8 (halves uniform)
    return v;
}
__device__ __forceinline__ float rowsum16(float v) {
    v += xdpp<0xB1>(v);
    v += xdpp<0x4E>(v);
    v += xdpp<0x141>(v);
    v += xdpp<0x140>(v);
    return v;
}
#else
__device__ __forceinline__ float rowmax16(float v) {
    for (int m = 8; m >= 1; m >>= 1) v = fmaxf(v, __shfl_xor(v, m, 32));
    return v;
}
__device__ __forceinline__ float rowsum16(float v) {
    for (int m = 8; m >= 1; m >>= 1) v += __shfl_xor(v, m, 32);
    return v;
}
#endif

// ---- async global -> LDS (ASYNCcnt) ---------------------------------------
#if defined(__gfx1250__) && __has_builtin(__builtin_amdgcn_global_load_async_to_lds_b128)
#define HAVE_ASYNC 1
typedef __attribute__((__vector_size__(4 * sizeof(int)))) int vsi4;
typedef __attribute__((address_space(1))) vsi4 gvsi4;
typedef __attribute__((address_space(3))) vsi4 lvsi4;
__device__ __forceinline__ void async_copy16(const void* g, void* l) {
    __builtin_amdgcn_global_load_async_to_lds_b128(
        (gvsi4*)(unsigned long long)(size_t)g,
        (lvsi4*)(unsigned)(size_t)l, 0, 0);
}
#if __has_builtin(__builtin_amdgcn_s_wait_asynccnt)
#define WAIT_ASYNC() __builtin_amdgcn_s_wait_asynccnt(0)
#else
#define WAIT_ASYNC() asm volatile("s_wait_asynccnt 0x0" ::: "memory")
#endif
#else
#define HAVE_ASYNC 0
#endif

// ---- Tensor Data Mover: 2D bf16 tile -> LDS with row padding ---------------
#if defined(__gfx1250__) && __has_builtin(__builtin_amdgcn_tensor_load_to_lds) && \
    __has_builtin(__builtin_amdgcn_s_wait_tensorcnt)
#define HAVE_TDM 1
typedef __attribute__((ext_vector_type(4))) unsigned int v4u_;
typedef __attribute__((ext_vector_type(8))) int v8i_;
typedef __attribute__((ext_vector_type(4))) int v4i_;
// D# per CDNA5 ISA ch.8: group0 = {count/lds_addr/global_addr/type},
// group1 = {data_size/pad/tensor dims/tile dims/strides}. 2D: groups 2,3 zero.
// Toolchain builtin is the 6-arg flavor: (g0, g1, g2, g3, g4, cpol).
__device__ __forceinline__ void tdm_load_2d_bf16(
        const void* gsrc, void* ldst,
        unsigned tensor_w, unsigned tensor_h,
        unsigned tile_w, unsigned tile_h,
        unsigned stride_elems,
        unsigned pad_interval_code, unsigned pad_amount_code, unsigned pad_en) {
    unsigned long long ga = (unsigned long long)(size_t)gsrc;
    unsigned la = (unsigned)(size_t)ldst;
    v4u_ g0;
    g0[0] = 1u;                                              // count=1 (user D#)
    g0[1] = la;                                              // lds_addr (bytes)
    g0[2] = (unsigned)ga;                                    // global_addr[31:0]
    g0[3] = (unsigned)((ga >> 32) & 0x01FFFFFFu) | (2u << 30); // addr[56:32], type=2
    v8i_ g1;
    g1[0] = (int)((1u << 16) |                               // data_size: 1 => 2B
                  (pad_en << 20) |                           // pad_enable
                  (pad_interval_code << 22) |                // DWORDs before pad
                  (pad_amount_code << 25));                  // pad DWORDs
    g1[1] = (int)((tensor_w & 0xFFFFu) << 16);               // tensor_dim0[15:0]
    g1[2] = (int)((tensor_w >> 16) | ((tensor_h & 0xFFFFu) << 16));
    g1[3] = (int)((tensor_h >> 16) | (tile_w << 16));        // tile_dim0
    g1[4] = (int)tile_h;                                     // tile_dim1, tile_dim2=0
    g1[5] = (int)stride_elems;                               // dim0_stride[31:0]
    g1[6] = 0;
    g1[7] = 0;
    v4i_ z4 = {0, 0, 0, 0};
    v8i_ z8 = {0, 0, 0, 0, 0, 0, 0, 0};
    __builtin_amdgcn_tensor_load_to_lds(g0, g1, z4, z4, z8, 0);
}
#define WAIT_TENSOR() __builtin_amdgcn_s_wait_tensorcnt(0)
#else
#define HAVE_TDM 0
#endif

#define HID   1024
#define NH    16
#define HD    64
#define SEQ   2048
#define BATCH 4

// ---------------------------------------------------------------------------
// Kernel 1: RMSNorm, one wave (32 lanes) per row of 1024, fp32 -> bf16
// ---------------------------------------------------------------------------
__global__ __launch_bounds__(256)
void rmsnorm_kernel(const float* __restrict__ x, const float* __restrict__ w,
                    unsigned short* __restrict__ xn) {
    int wave = threadIdx.x >> 5, lane = threadIdx.x & 31;
    int row  = blockIdx.x * 8 + wave;
    const float* xr = x + (size_t)row * HID;

    float4 vals[8];
    float  ss = 0.f;
#pragma unroll
    for (int i = 0; i < 8; ++i) {
        vals[i] = *(const float4*)(xr + lane * 4 + i * 128);
        ss += vals[i].x * vals[i].x + vals[i].y * vals[i].y +
              vals[i].z * vals[i].z + vals[i].w * vals[i].w;
    }
    ss = rowsum16(ss);
    ss += __shfl_xor(ss, 16, 32);   // cross-half step
    float r = rsqrtf(ss * (1.0f / (float)HID) + 1e-5f);

    unsigned short* orow = xn + (size_t)row * HID;
#pragma unroll
    for (int i = 0; i < 8; ++i) {
        float4 wv = *(const float4*)(w + lane * 4 + i * 128);
        uint2 p;
        p.x = pack2(vals[i].x * r * wv.x, vals[i].y * r * wv.y);
        p.y = pack2(vals[i].z * r * wv.z, vals[i].w * r * wv.w);
        *(uint2*)(orow + lane * 4 + i * 128) = p;
    }
}

// ---------------------------------------------------------------------------
// Kernel 2/4: GEMM  C[m,n] = sum_k A[m,k]*W[n,k] + bias[n]
//   A : bf16 [M x K] (row major),  W : fp32 [N x K] (row major, native layout)
//   MODE 0: store bf16, MODE 1: store fp32
//   Block 256 thr = 8 waves, tile 128x128, k-chunk 32; wave tile 32x64.
//   A tile arrives by TDM DMA (pad_interval 16 DW + pad 4 DW == 80B LDS rows).
// ---------------------------------------------------------------------------
template <int MODE>
__global__ __launch_bounds__(256)
void gemm_kernel(const unsigned short* __restrict__ A,
                 const float* __restrict__ W,
                 const float* __restrict__ bias,
                 void* __restrict__ Cout, int N, int K) {
    __shared__ unsigned short As[128][40];   // [m][k]   (pad 8 => 80B rows)
    __shared__ unsigned short Bs[128][40];   // [n][k]   (pad 8)

    int tid  = threadIdx.x;
    int wave = tid >> 5, lane = tid & 31;
    int bm   = blockIdx.x * 128, bn = blockIdx.y * 128;
    int wm   = (wave & 3) * 32, wn = (wave >> 2) * 64;

    f32x8 acc[2][4] = {};

    int srow = tid >> 1;
    int koff = (tid & 1) * 16;

    for (int k0 = 0; k0 < K; k0 += 32) {
        __syncthreads();
        // stage A (straight bf16 copy): TDM descriptor DMA > async DMA > manual
#if HAVE_TDM
        if (wave == 0)
            tdm_load_2d_bf16(A + (size_t)bm * K + k0, &As[0][0],
                             /*tensor_w=*/(unsigned)K, /*tensor_h=*/1u << 24,
                             /*tile_w=*/32, /*tile_h=*/128,
                             /*stride=*/(unsigned)K,
                             /*pad_interval(16DW)=*/3, /*pad_amount(4DW)=*/3, 1);
#elif HAVE_ASYNC
        {
            const unsigned short* src = A + (size_t)(bm + srow) * K + k0 + koff;
            async_copy16(src,     &As[srow][koff]);
            async_copy16(src + 8, &As[srow][koff + 8]);
        }
#else
        {
            const uint4* s4 = (const uint4*)(A + (size_t)(bm + srow) * K + k0 + koff);
            *(uint4*)&As[srow][koff]     = s4[0];
            *(uint4*)&As[srow][koff + 8] = s4[1];
        }
#endif
        // stage W (fp32 -> bf16 on the fly, row major: lane-contiguous B frags)
        {
            const float4* src = (const float4*)(W + (size_t)(bn + srow) * K + k0 + koff);
            float4 f0 = src[0], f1 = src[1], f2 = src[2], f3 = src[3];
            uint4 lo, hi;
            lo.x = pack2(f0.x, f0.y); lo.y = pack2(f0.z, f0.w);
            lo.z = pack2(f1.x, f1.y); lo.w = pack2(f1.z, f1.w);
            hi.x = pack2(f2.x, f2.y); hi.y = pack2(f2.z, f2.w);
            hi.z = pack2(f3.x, f3.y); hi.w = pack2(f3.z, f3.w);
            *(uint4*)&Bs[srow][koff]     = lo;
            *(uint4*)&Bs[srow][koff + 8] = hi;
        }
        if (k0 + 32 < K)
            __builtin_prefetch(W + (size_t)(bn + srow) * K + k0 + 32 + koff, 0, 1);
#if HAVE_TDM
        if (wave == 0) WAIT_TENSOR();
#elif HAVE_ASYNC
        WAIT_ASYNC();
#endif
        __syncthreads();

        // fragments
        FragAB afr[2], bfr[4];
        int ah = (lane >> 4) * 16;          // A half: K 0..15 / 16..31
#pragma unroll
        for (int i = 0; i < 2; ++i) {
            const unsigned short* p = &As[wm + 16 * i + (lane & 15)][0];
            afr[i].q[0] = *(const uint4*)(p + ah);
            afr[i].q[1] = *(const uint4*)(p + 8 + ah);
        }
        int bh = (lane >> 4) * 16;          // B half: K 0..15 / 16..31
#pragma unroll
        for (int j = 0; j < 4; ++j) {
            const unsigned short* p = &Bs[wn + 16 * j + (lane & 15)][bh];
            bfr[j].q[0] = *(const uint4*)(p);
            bfr[j].q[1] = *(const uint4*)(p + 8);
        }
#pragma unroll
        for (int i = 0; i < 2; ++i)
#pragma unroll
            for (int j = 0; j < 4; ++j)
                acc[i][j] = wmma_bf16(afr[i].v, bfr[j].v, acc[i][j]);
    }

    // epilogue: C layout lane l, vgpr r -> M = r + 8*(l>>4), N = l&15
    int chalf = (lane >> 4) * 8;
#pragma unroll
    for (int j = 0; j < 4; ++j) {
        int n = bn + wn + 16 * j + (lane & 15);
        float bv = bias[n];
#pragma unroll
        for (int i = 0; i < 2; ++i) {
#pragma unroll
            for (int r = 0; r < 8; ++r) {
                int m = bm + wm + 16 * i + r + chalf;
                float v = acc[i][j][r] + bv;
                if (MODE == 0)
                    ((unsigned short*)Cout)[(size_t)m * N + n] = f2bf(v);
                else
                    ((float*)Cout)[(size_t)m * N + n] = v;
            }
        }
    }
}

// ---------------------------------------------------------------------------
// Kernel 3: flash attention. Block = 128 thr (4 waves), one (b,h, 64 q-rows).
// Each wave owns 16 query rows; key/value blocks of 32.
// qkv layout: bf16 [B*S][3*HID], Q at col h*64, K at 1024+h*64, V at 2048+h*64
// ---------------------------------------------------------------------------
__global__ __launch_bounds__(128)
void attn_kernel(const unsigned short* __restrict__ qkv,
                 unsigned short* __restrict__ obuf) {
    __shared__ unsigned short Ks[32][72];      // [key][d]   (pad 8)
    __shared__ unsigned short Vt[64][40];      // [d][key]   transposed (pad 8)
    __shared__ unsigned short Ps[4][16][40];   // per-wave P tile 16x32 (pad 8)

    const int LDQ = 3 * HID;
    int tid = threadIdx.x, wave = tid >> 5, lane = tid & 31;
    int idx = blockIdx.x;
    int b  = idx >> 9;          // 16 heads * 32 q-tiles
    int h  = (idx >> 5) & 15;
    int qt = idx & 31;
    size_t baseRow = (size_t)b * SEQ;
    int q0 = qt * 64 + wave * 16;

    // Q A-fragments (16 rows x 64 d -> two 16x32 frags), loaded once
    FragAB qa[2];
    {
        int mrow = q0 + (lane & 15);
        const unsigned short* qptr = qkv + (baseRow + mrow) * LDQ + h * HD;
        int ah = (lane >> 4) * 16;
#pragma unroll
        for (int c = 0; c < 2; ++c) {
            qa[c].q[0] = *(const uint4*)(qptr + 32 * c + ah);
            qa[c].q[1] = *(const uint4*)(qptr + 32 * c + 8 + ah);
        }
    }

    f32x8 o[4] = {};
    float mrun[8], lrun[8];
#pragma unroll
    for (int r = 0; r < 8; ++r) { mrun[r] = -1e30f; lrun[r] = 0.f; }

    for (int j0 = 0; j0 < SEQ; j0 += 32) {
        __syncthreads();
        int key = tid >> 2, doff = (tid & 3) * 16;
        // stage K rows (straight copy): QK^T B-frags read contiguous rows
        {
            const unsigned short* src = qkv + (baseRow + j0 + key) * LDQ +
                                        HID + h * HD + doff;
#if HAVE_ASYNC
            async_copy16(src,     &Ks[key][doff]);
            async_copy16(src + 8, &Ks[key][doff + 8]);
#else
            const uint4* s4 = (const uint4*)src;
            *(uint4*)&Ks[key][doff]     = s4[0];
            *(uint4*)&Ks[key][doff + 8] = s4[1];
#endif
        }
        // stage V transposed: P*V B-frags read contiguous [d][key] rows
        {
            const unsigned short* src = qkv + (baseRow + j0 + key) * LDQ +
                                        2 * HID + h * HD + doff;
            uint4 d0 = ((const uint4*)src)[0], d1 = ((const uint4*)src)[1];
            unsigned short tmp[16];
            *(uint4*)tmp       = d0;
            *(uint4*)(tmp + 8) = d1;
#pragma unroll
            for (int i = 0; i < 16; ++i) Vt[doff + i][key] = tmp[i];
        }
#if HAVE_ASYNC
        WAIT_ASYNC();
#endif
        __syncthreads();

        // scores: two 16x16 frags (keys g*16..g*16+15), K-dim = d (2 chunks)
        f32x8 s[2];
#pragma unroll
        for (int g = 0; g < 2; ++g) {
            int krow = g * 16 + (lane & 15);
            int dsel = (lane >> 4) * 16;
            FragAB kb0, kb1;
            kb0.q[0] = *(const uint4*)&Ks[krow][dsel];
            kb0.q[1] = *(const uint4*)&Ks[krow][dsel + 8];
            kb1.q[0] = *(const uint4*)&Ks[krow][32 + dsel];
            kb1.q[1] = *(const uint4*)&Ks[krow][32 + dsel + 8];
            f32x8 z = {};
            z = wmma_bf16(qa[0].v, kb0.v, z);
            z = wmma_bf16(qa[1].v, kb1.v, z);
            s[g] = z;
        }

        // online softmax (row = vgpr index, cols striped over 16-lane halves)
        const float sc = 0.125f;   // 1/sqrt(64)
        float corr[8];
#pragma unroll
        for (int r = 0; r < 8; ++r) {
            float s0 = s[0][r] * sc, s1 = s[1][r] * sc;
            float mx = rowmax16(fmaxf(s0, s1));
            float mnew = fmaxf(mrun[r], mx);
            float c  = __expf(mrun[r] - mnew);
            float p0 = __expf(s0 - mnew);
            float p1 = __expf(s1 - mnew);
            float rs = rowsum16(p0 + p1);
            lrun[r] = lrun[r] * c + rs;
            mrun[r] = mnew;
            corr[r] = c;
            s[0][r] = p0;
            s[1][r] = p1;
        }
#pragma unroll
        for (int t = 0; t < 4; ++t)
#pragma unroll
            for (int r = 0; r < 8; ++r) o[t][r] *= corr[r];

        // P: C-layout -> LDS -> A-layout (per-wave tile)
#pragma unroll
        for (int r = 0; r < 8; ++r) {
            int row = r + 8 * (lane >> 4);
            Ps[wave][row][lane & 15]        = f2bf(s[0][r]);
            Ps[wave][row][16 + (lane & 15)] = f2bf(s[1][r]);
        }
        __syncthreads();

        FragAB pa;
        {
            int prow = lane & 15;
            int ksel = (lane >> 4) * 16;
            pa.q[0] = *(const uint4*)&Ps[wave][prow][ksel];
            pa.q[1] = *(const uint4*)&Ps[wave][prow][8 + ksel];
        }
#pragma unroll
        for (int t = 0; t < 4; ++t) {
            FragAB vb;
            int drow = t * 16 + (lane & 15);
            int ks2  = (lane >> 4) * 16;
            vb.q[0] = *(const uint4*)&Vt[drow][ks2];
            vb.q[1] = *(const uint4*)&Vt[drow][ks2 + 8];
            o[t] = wmma_bf16(pa.v, vb.v, o[t]);
        }
    }

    // normalize + store O (bf16) at [row][h*64 + d]
#pragma unroll
    for (int r = 0; r < 8; ++r) {
        float inv = 1.0f / lrun[r];
        int row = q0 + r + 8 * (lane >> 4);
        unsigned short* dst = obuf + (baseRow + row) * HID + h * HD + (lane & 15);
#pragma unroll
        for (int t = 0; t < 4; ++t) dst[t * 16] = f2bf(o[t][r] * inv);
    }
}

// ---------------------------------------------------------------------------
// launch
// ---------------------------------------------------------------------------
extern "C" void kernel_launch(void* const* d_in, const int* in_sizes, int n_in,
                              void* d_out, int out_size, void* d_ws, size_t ws_size,
                              hipStream_t stream) {
    const float* x      = (const float*)d_in[0];
    const float* w_in   = (const float*)d_in[1];
    const float* b_in   = (const float*)d_in[2];
    const float* w_norm = (const float*)d_in[3];
    const float* w_out  = (const float*)d_in[4];
    const float* b_out  = (const float*)d_in[5];
    float* out = (float*)d_out;

    const size_t M = (size_t)BATCH * SEQ;                 // 8192
    char* ws = (char*)d_ws;
    unsigned short* xn   = (unsigned short*)ws;                               // 16 MB
    unsigned short* qkv  = (unsigned short*)(ws + M * HID * 2);               // 50 MB
    unsigned short* obuf = (unsigned short*)(ws + M * HID * 2 + M * 3 * HID * 2);

    rmsnorm_kernel<<<(int)(M / 8), 256, 0, stream>>>(x, w_norm, xn);

    gemm_kernel<0><<<dim3((int)(M / 128), 3 * HID / 128), 256, 0, stream>>>(
        xn, w_in, b_in, (void*)qkv, 3 * HID, HID);

    attn_kernel<<<BATCH * NH * (SEQ / 64), 128, 0, stream>>>(qkv, obuf);

    gemm_kernel<1><<<dim3((int)(M / 128), HID / 128), 256, 0, stream>>>(
        obuf, w_out, b_out, (void*)out, HID, HID);
}
